// ArcLoss_2516850836126
// MI455X (gfx1250) — matrix operand order repaired and checked by reading, not compiled, not run on hardware
//
#include <hip/hip_runtime.h>

typedef __attribute__((ext_vector_type(16))) _Float16 v16h;
typedef __attribute__((ext_vector_type(8)))  _Float16 v8h;
typedef __attribute__((ext_vector_type(8)))  float    v8f;
typedef __attribute__((ext_vector_type(4)))  int      v4i;

#define MDIM 2048
#define KDIM 128
#define NDIM 50000
#define NPAD 50048          // 391 * 128
#define NBLK 391
#define COS_M 0.98006657784124163f
#define TEN_SIN_M 1.9866933079506122f   // 10 * sin(0.2)

#if defined(__AMDGCN__) && __has_builtin(__builtin_amdgcn_global_load_async_to_lds_b128) && __has_builtin(__builtin_amdgcn_s_wait_asynccnt)
#define ASYNC_STAGE 1
typedef __attribute__((address_space(1))) v4i gv4i;
typedef __attribute__((address_space(3))) v4i lv4i;
#else
#define ASYNC_STAGE 0
#endif

// ---------------- x row-normalize -> f16 [2048][128] ----------------
__global__ __launch_bounds__(256) void normalize_x(const float* __restrict__ x,
                                                   _Float16* __restrict__ xn) {
    const int wave = threadIdx.x >> 5;
    const int lane = threadIdx.x & 31;
    const int row  = blockIdx.x * 8 + wave;           // 256 blocks * 8 rows
    const float* xr = x + (size_t)row * KDIM + lane * 4;
    float v0 = xr[0], v1 = xr[1], v2 = xr[2], v3 = xr[3];
    float ss = v0 * v0 + v1 * v1 + v2 * v2 + v3 * v3;
#pragma unroll
    for (int m = 16; m >= 1; m >>= 1) ss += __shfl_xor(ss, m, 32);
    const float rn = rsqrtf(ss);
    _Float16* o = xn + (size_t)row * KDIM + lane * 4;
    o[0] = (_Float16)(v0 * rn);
    o[1] = (_Float16)(v1 * rn);
    o[2] = (_Float16)(v2 * rn);
    o[3] = (_Float16)(v3 * rn);
}

// ---------------- w column-normalize -> f16 w^T [NPAD][128] ----------------
__global__ __launch_bounds__(256) void normalize_w(const float* __restrict__ w,
                                                   _Float16* __restrict__ wnT) {
    const int col = blockIdx.x * 256 + threadIdx.x;
    if (col >= NPAD) return;
    _Float16* o = wnT + (size_t)col * KDIM;
    if (col < NDIM) {
        float ss = 0.f;
#pragma unroll 8
        for (int k = 0; k < KDIM; ++k) {
            float v = w[(size_t)k * NDIM + col];
            ss += v * v;
        }
        const float rn = rsqrtf(ss);
#pragma unroll 8
        for (int k = 0; k < KDIM; ++k)
            o[k] = (_Float16)(w[(size_t)k * NDIM + col] * rn);
    } else {
#pragma unroll 8
        for (int k = 0; k < KDIM; ++k) o[k] = (_Float16)0.f;
    }
}

// ---------------- fused WMMA GEMM (d = xn @ wnT^T), two modes ----------------
// SUMPASS:  writes per-(row, colblock) partial sums of exp(d) -> partial[blk*M + row]
// !SUMPASS: writes out = g - log(exp(g) + S_row - exp(d))
template <bool SUMPASS>
__global__ __launch_bounds__(256) void arc_gemm(const _Float16* __restrict__ xn,
                                                const _Float16* __restrict__ wnT,
                                                const float* __restrict__ S,
                                                float* __restrict__ dst) {
    __shared__ _Float16 Bs[128 * KDIM];               // 32 KB: 128 cols x 128 K
    const int tid  = threadIdx.x;
    const int wave = tid >> 5;
    const int lane = tid & 31;
    const int half = lane >> 4;                        // 0: lanes 0-15, 1: 16-31
    const int l16  = lane & 15;
    const int colBase = blockIdx.x * 128;
    const int rowBase = blockIdx.y * 128 + wave * 16;  // wave owns 16 rows

    // Stage B tile: 32 KB, 128 B per thread, coalesced 16 B chunks
#if ASYNC_STAGE
    {
        const char* srcB = (const char*)(wnT + (size_t)colBase * KDIM);
        char* dstB = (char*)Bs;
#pragma unroll
        for (int i = 0; i < 8; ++i) {
            const int c = (tid + i * 256) * 16;
            __builtin_amdgcn_global_load_async_to_lds_b128(
                (gv4i*)(srcB + c), (lv4i*)(dstB + c), 0, 0);
        }
        __builtin_amdgcn_s_wait_asynccnt(0);
    }
#else
    {
        const v8h* __restrict__ src = (const v8h*)(wnT + (size_t)colBase * KDIM);
        v8h* dstS = (v8h*)Bs;
#pragma unroll
        for (int i = 0; i < 8; ++i) {
            const int c = tid + i * 256;
            dstS[c] = src[c];
        }
    }
#endif
    __syncthreads();

    v8f acc[8];
#pragma unroll
    for (int t = 0; t < 8; ++t) acc[t] = (v8f)0.f;

#pragma unroll
    for (int ks = 0; ks < 4; ++ks) {
        const int k0 = ks * 32;
        // A fragment 16x32 f16: lane<16 -> K[k0..k0+15]; lane>=16 -> K[k0+16..k0+31]
        const _Float16* ap = xn + (size_t)(rowBase + l16) * KDIM + k0 + half * 16;
        const v8h alo = *(const v8h*)(ap);
        const v8h ahi = *(const v8h*)(ap + 8);
        v16h a;
#pragma unroll
        for (int i = 0; i < 8; ++i) { a[i] = alo[i]; a[8 + i] = ahi[i]; }
#pragma unroll
        for (int t = 0; t < 8; ++t) {
            // B fragment 32x16: lane holds 16 contiguous K of one column
            const _Float16* bp = Bs + (size_t)(t * 16 + l16) * KDIM + k0 + half * 16;
            const v16h b = *(const v16h*)bp;
            acc[t] = __builtin_amdgcn_wmma_f32_16x16x32_f16(
                false, a, false, b, (short)0, acc[t], false, false);
        }
    }

    if (SUMPASS) {
        float p[8];
#pragma unroll
        for (int r = 0; r < 8; ++r) p[r] = 0.f;
#pragma unroll
        for (int t = 0; t < 8; ++t) {
            const int col = colBase + t * 16 + l16;
            const bool valid = col < NDIM;
#pragma unroll
            for (int r = 0; r < 8; ++r)
                p[r] += valid ? __expf(acc[t][r]) : 0.f;
        }
        // reduce over the 16 lanes of each half (columns); rows stay per-half
#pragma unroll
        for (int m = 1; m < 16; m <<= 1) {
#pragma unroll
            for (int r = 0; r < 8; ++r) p[r] += __shfl_xor(p[r], m, 32);
        }
        if (l16 == 0) {
            float* part = dst + (size_t)blockIdx.x * MDIM + rowBase + half * 8;
#pragma unroll
            for (int r = 0; r < 8; ++r) part[r] = p[r];
        }
    } else {
        float sv[8];
#pragma unroll
        for (int r = 0; r < 8; ++r) sv[r] = S[rowBase + half * 8 + r];
#pragma unroll
        for (int t = 0; t < 8; ++t) {
            const int col = colBase + t * 16 + l16;
            if (col < NDIM) {
#pragma unroll
                for (int r = 0; r < 8; ++r) {
                    const float d = acc[t][r];
                    const float c = d * 0.1f;
                    // sqrt(1-c^2) via even polynomial; |c| <= ~0.1 so rel.err < 5e-10
                    const float u = c * c;
                    const float s = 1.0f - u * (0.5f + u * (0.125f + u * 0.0625f));
                    const float g = fmaf(d, COS_M, -TEN_SIN_M * s);
                    const float val = g - __logf(__expf(g) + sv[r] - __expf(d));
                    dst[(size_t)(rowBase + half * 8 + r) * NDIM + col] = val;
                }
            }
        }
    }
}

// ---------------- deterministic reduction of partials -> S[2048] ----------------
__global__ __launch_bounds__(256) void reduce_S(const float* __restrict__ partial,
                                                float* __restrict__ S) {
    const int row = blockIdx.x * 256 + threadIdx.x;    // 8 blocks * 256
    float s = 0.f;
    for (int j = 0; j < NBLK; ++j) s += partial[(size_t)j * MDIM + row];
    S[row] = s;
}

extern "C" void kernel_launch(void* const* d_in, const int* in_sizes, int n_in,
                              void* d_out, int out_size, void* d_ws, size_t ws_size,
                              hipStream_t stream) {
    const float* x = (const float*)d_in[0];   // [2048][128]
    const float* w = (const float*)d_in[1];   // [128][50000]
    float* out = (float*)d_out;               // [2048][50000]

    char* ws = (char*)d_ws;
    const size_t xn_bytes   = (size_t)MDIM * KDIM * sizeof(_Float16);      // 512 KB
    const size_t wnT_bytes  = (size_t)NPAD * KDIM * sizeof(_Float16);      // ~12.8 MB
    const size_t part_bytes = (size_t)NBLK * MDIM * sizeof(float);         // ~3.2 MB
    size_t off = 0;
    _Float16* xn  = (_Float16*)(ws + off); off += (xn_bytes + 255) & ~(size_t)255;
    _Float16* wnT = (_Float16*)(ws + off); off += (wnT_bytes + 255) & ~(size_t)255;
    float* partial = (float*)(ws + off);   off += (part_bytes + 255) & ~(size_t)255;
    float* S       = (float*)(ws + off);

    normalize_x<<<MDIM / 8, 256, 0, stream>>>(x, xn);
    normalize_w<<<(NPAD + 255) / 256, 256, 0, stream>>>(w, wnT);

    dim3 grid(NBLK, MDIM / 128);
    arc_gemm<true><<<grid, 256, 0, stream>>>(xn, wnT, nullptr, partial);
    reduce_S<<<MDIM / 256, 256, 0, stream>>>(partial, S);
    arc_gemm<false><<<grid, 256, 0, stream>>>(xn, wnT, S, out);
}
